// CT_AUG_system_70042326663203
// MI455X (gfx1250) — matrix operand order repaired and checked by reading, not compiled, not run on hardware
//
#include <hip/hip_runtime.h>
#include <hip/hip_bf16.h>
#include <math.h>

// Problem sizes (fixed by the reference)
#define NN   512
#define MM   512
#define DD   1024
#define H1   512
#define H2   256

typedef __attribute__((ext_vector_type(16))) __bf16 v16bf;
typedef __attribute__((ext_vector_type(8)))  float  v8f;

union frag_u { v16bf v; uint4 q[2]; __bf16 e[16]; };

// ---------------------------------------------------------------------------
// CDNA5 WMMA fragment layouts (wave32):
//   A (16x32 bf16): lane L -> row M=L%16; elems e=0..7 -> K=8*(L/16)+e,
//                   e=8..15 -> K=16+8*(L/16)+(e-8)
//   B (32x16 bf16): lane L -> col N=L%16; elem e -> K as for A (8*(L/16)+e, then 16+8*(L/16)+(e-8))
//   C/D (16x16 f32): lane L, reg r -> row=r+8*(L/16), col=L%16
// Weights are pre-swizzled in global memory so that a lane's B elements are
// two 16-byte runs 32 bytes apart:  Wp[kblock][col][kk]  (kk = 0..31 k-major, bf16).
// ---------------------------------------------------------------------------
static __device__ inline v16bf load_a_frag_lds(const __bf16* A, int ldA, int lane) {
    int m = lane & 15, h = lane >> 4;
    frag_u u;
    const uint4* p = reinterpret_cast<const uint4*>(A + m * ldA + 8 * h);
    u.q[0] = p[0];        // K = 8h .. 8h+7
    u.q[1] = p[2];        // K = 16+8h .. +7  (+32 bytes)
    return u.v;
}

static __device__ inline v16bf load_b_frag_g(const __bf16* Wp) {
    frag_u u;
    const uint4* p = reinterpret_cast<const uint4*>(Wp);
    u.q[0] = p[0];
    u.q[1] = p[2];
    return u.v;
}

static __device__ inline v8f wmma_bf16(v16bf a, v16bf b, v8f c) {
    return __builtin_amdgcn_wmma_f32_16x16x32_bf16(
        false, a, false, b, (short)0, c, false, false);
}

static __device__ inline float leaky(float x) {
    return x >= 0.0f ? x : 0.01f * x;
}

// ---------------------------------------------------------------------------
// Kernel 0: convert W1/W2 to bf16 AND swizzle into B-fragment layout:
//   Wp[(k/32) * (C*32) + c*32 + (k%32)] = W[k*C + c]
// ---------------------------------------------------------------------------
__global__ void swizzle_weights(const float* __restrict__ W1,
                                const float* __restrict__ W2,
                                __bf16* __restrict__ W1p,
                                __bf16* __restrict__ W2p) {
    int idx = blockIdx.x * blockDim.x + threadIdx.x;
    if (idx < DD * H1) {
        int c = idx & (H1 - 1);
        int k = idx >> 9;
        W1p[(k >> 5) * (H1 * 32) + c * 32 + (k & 31)] = (__bf16)W1[idx];
    }
    int j = idx - DD * H1;
    if (j >= 0 && j < H1 * H2) {
        int c = j & (H2 - 1);
        int k = j >> 8;
        W2p[(k >> 5) * (H2 * 32) + c * 32 + (k & 31)] = (__bf16)W2[j];
    }
}

// ---------------------------------------------------------------------------
// Kernel 1: fused mse -> MLP(1024->512->256->1) per (16n x 4m) tile.
// 256 threads = 8 waves; wave w: row-tile rt=w&3 (m = m0+rt), col half w>>2.
// Layer-1 k-loop is barrier-free: A fragments built in registers from
// M_org/Q, B fragments loaded directly from pre-swizzled weights (L2).
// LDS: h1 (64x512 bf16, 64KB) + h2 (64x256 bf16, 32KB) = 96KB dynamic.
// ---------------------------------------------------------------------------
__global__ __launch_bounds__(256, 1)
void fused_mlp_kernel(const float* __restrict__ Morg,
                      const float* __restrict__ Qm,
                      const __bf16* __restrict__ W1p,
                      const float* __restrict__ b1,
                      const __bf16* __restrict__ W2p,
                      const float* __restrict__ b2,
                      const float* __restrict__ W3,
                      const float* __restrict__ b3,
                      float* __restrict__ dT,      // [MM][NN]
                      float* __restrict__ costT) { // [MM][NN]
    extern __shared__ char smem[];
    __bf16* h1S = (__bf16*)(smem);            // 64 x 512
    __bf16* h2S = (__bf16*)(smem + 65536);    // 64 x 256

    const int tid   = threadIdx.x;
    const int lane  = tid & 31;
    const int wv    = tid >> 5;               // 0..7
    const int m0    = (blockIdx.x & 127) * 4;
    const int n0    = (blockIdx.x >> 7) * 16;

    const int rt     = wv & 3;                // row tile (== local m)
    const int rowOff = rt * 16;
    const int cBase1 = (wv >> 2) * 256;       // layer1: 16 col tiles
    const int cBase2 = (wv >> 2) * 128;       // layer2: 8 col tiles
    const int halfL  = lane >> 4;
    const int colL   = lane & 15;

    // ---------------- Layer 1: h1 = leaky(mse @ W1 + b1) ----------------
    v8f acc1[16];
    #pragma unroll
    for (int t = 0; t < 16; ++t)
        #pragma unroll
        for (int r = 0; r < 8; ++r) acc1[t][r] = 0.0f;

    const int   nLane = n0 + colL;            // this lane's n (A row)
    const int   mWave = m0 + rt;              // this wave's m
    const float* mp = Morg + nLane * DD;
    const float* qp = Qm   + mWave * DD;

    // per-lane B base offset within a k-block (elements)
    const int bOff1 = (cBase1 + colL) * 32 + 8 * halfL;
    const int bOff2 = (cBase2 + colL) * 32 + 8 * halfL;

    float costAcc = 0.0f;

    for (int kb = 0; kb < DD / 32; ++kb) {
        const int k0 = kb * 32;
        // ---- build A fragment in registers: mse = (M - Q)^2, bf16 ----
        frag_u u;
        {
            const float4* ma = (const float4*)(mp + k0 + 8 * halfL);
            const float4* qa = (const float4*)(qp + k0 + 8 * halfL);
            float4 a0 = ma[0], a1 = ma[1], a2 = ma[4], a3 = ma[5];
            float4 q0 = qa[0], q1 = qa[1], q2 = qa[4], q3 = qa[5];
            float d0, d1, d2, d3;
            #define MSE4(av, qv, e0)                                        \
                d0 = av.x - qv.x; d1 = av.y - qv.y;                         \
                d2 = av.z - qv.z; d3 = av.w - qv.w;                         \
                d0 *= d0; d1 *= d1; d2 *= d2; d3 *= d3;                     \
                costAcc += d0 + d1 + d2 + d3;                               \
                u.e[e0 + 0] = (__bf16)d0; u.e[e0 + 1] = (__bf16)d1;         \
                u.e[e0 + 2] = (__bf16)d2; u.e[e0 + 3] = (__bf16)d3;
            MSE4(a0, q0, 0)
            MSE4(a1, q1, 4)
            MSE4(a2, q2, 8)
            MSE4(a3, q3, 12)
            #undef MSE4
        }
        // ---- 16 WMMAs, B straight from L2-resident swizzled W1 ----
        const __bf16* wbase = W1p + kb * (H1 * 32) + bOff1;
        #pragma unroll
        for (int ct = 0; ct < 16; ++ct) {
            v16bf bf = load_b_frag_g(wbase + ct * (16 * 32));
            acc1[ct] = wmma_bf16(u.v, bf, acc1[ct]);
        }
        // prefetch next k-block of this wave's W1 columns
        if (kb + 1 < DD / 32)
            __builtin_prefetch(W1p + (kb + 1) * (H1 * 32) + cBase1 * 32 + lane * 256, 0, 0);
    }

    // exact f32 cost: lane L holds half the row sum; L ^ 16 holds the rest
    {
        float v = costAcc + __shfl_xor(costAcc, 16, 32);
        if (wv < 4 && lane < 16)
            costT[mWave * NN + n0 + lane] = v;
    }

    // epilogue: bias + leaky relu -> h1 (bf16 LDS)
    #pragma unroll
    for (int ct = 0; ct < 16; ++ct) {
        int c = cBase1 + ct * 16 + colL;
        float bias = b1[c];
        #pragma unroll
        for (int r = 0; r < 8; ++r) {
            float x = leaky(acc1[ct][r] + bias);
            h1S[(rowOff + r + 8 * halfL) * H1 + c] = (__bf16)x;
        }
    }
    __syncthreads();

    // ---------------- Layer 2: h2 = leaky(h1 @ W2 + b2) ----------------
    v8f acc2[8];
    #pragma unroll
    for (int t = 0; t < 8; ++t)
        #pragma unroll
        for (int r = 0; r < 8; ++r) acc2[t][r] = 0.0f;

    for (int kb = 0; kb < H1 / 32; ++kb) {
        v16bf afrag = load_a_frag_lds(h1S + rowOff * H1 + kb * 32, H1, lane);
        const __bf16* wbase = W2p + kb * (H2 * 32) + bOff2;
        #pragma unroll
        for (int ct = 0; ct < 8; ++ct) {
            v16bf bf = load_b_frag_g(wbase + ct * (16 * 32));
            acc2[ct] = wmma_bf16(afrag, bf, acc2[ct]);
        }
    }
    #pragma unroll
    for (int ct = 0; ct < 8; ++ct) {
        int c = cBase2 + ct * 16 + colL;
        float bias = b2[c];
        #pragma unroll
        for (int r = 0; r < 8; ++r) {
            float x = leaky(acc2[ct][r] + bias);
            h2S[(rowOff + r + 8 * halfL) * H2 + c] = (__bf16)x;
        }
    }
    __syncthreads();

    // ---------------- Layer 3: d = -(h2 @ W3 + b3) ----------------------
    if (tid < 64) {
        float s = 0.0f;
        #pragma unroll 8
        for (int j = 0; j < H2; ++j)
            s += (float)h2S[tid * H2 + j] * W3[j];
        float dv = -(s + b3[0]);
        int n = n0 + (tid & 15);
        int m = m0 + (tid >> 4);
        dT[m * NN + n] = dv;
    }
}

// ---------------------------------------------------------------------------
// Kernel 2: per-column softmax over n + weighted score. One WG per m.
// out[0..511] = score ; out[512 + n*512 + m] = backward_map[n][m]
// ---------------------------------------------------------------------------
__global__ void softmax_score_kernel(const float* __restrict__ dT,
                                     const float* __restrict__ costT,
                                     float* __restrict__ out) {
    __shared__ float red[256];
    const int m = blockIdx.x;
    const int t = threadIdx.x;

    float v0 = dT[m * NN + t];
    float v1 = dT[m * NN + 256 + t];

    red[t] = fmaxf(v0, v1);
    __syncthreads();
    for (int s = 128; s > 0; s >>= 1) {
        if (t < s) red[t] = fmaxf(red[t], red[t + s]);
        __syncthreads();
    }
    float mx = red[0];
    __syncthreads();

    float e0 = expf(v0 - mx);
    float e1 = expf(v1 - mx);
    red[t] = e0 + e1;
    __syncthreads();
    for (int s = 128; s > 0; s >>= 1) {
        if (t < s) red[t] += red[t + s];
        __syncthreads();
    }
    float inv = 1.0f / red[0];
    __syncthreads();

    float bm0 = e0 * inv;
    float bm1 = e1 * inv;
    out[512 + t * MM + m]         = bm0;
    out[512 + (t + 256) * MM + m] = bm1;

    red[t] = costT[m * NN + t] * bm0 + costT[m * NN + 256 + t] * bm1;
    __syncthreads();
    for (int s = 128; s > 0; s >>= 1) {
        if (t < s) red[t] += red[t + s];
        __syncthreads();
    }
    if (t == 0) out[m] = red[0];
}

// ---------------------------------------------------------------------------
extern "C" void kernel_launch(void* const* d_in, const int* in_sizes, int n_in,
                              void* d_out, int out_size, void* d_ws, size_t ws_size,
                              hipStream_t stream) {
    (void)in_sizes; (void)n_in; (void)out_size; (void)ws_size;

    const float* Morg = (const float*)d_in[0];
    const float* Qm   = (const float*)d_in[1];
    const float* W1   = (const float*)d_in[2];
    const float* b1   = (const float*)d_in[3];
    const float* W2   = (const float*)d_in[4];
    const float* b2   = (const float*)d_in[5];
    const float* W3   = (const float*)d_in[6];
    const float* b3   = (const float*)d_in[7];
    float* out        = (float*)d_out;

    // workspace layout (bytes)
    char* ws = (char*)d_ws;
    __bf16* W1p = (__bf16*)(ws + 0);          // 1024*512*2 = 1,048,576
    __bf16* W2p = (__bf16*)(ws + 1048576);    //  512*256*2 =   262,144
    float*  dT  = (float*) (ws + 1310720);    //  512*512*4 = 1,048,576
    float*  cT  = (float*) (ws + 2359296);    //  512*512*4 = 1,048,576

    // 1) weight conversion + swizzle into B-fragment layout
    swizzle_weights<<<(DD * H1 + H1 * H2 + 255) / 256, 256, 0, stream>>>(W1, W2, W1p, W2p);

    // 2) fused bf16-WMMA MLP over all (n,m) pairs
    const size_t smemBytes = 65536 + 32768;   // h1 + h2
    fused_mlp_kernel<<<(NN / 16) * (MM / 4), 256, smemBytes, stream>>>(
        Morg, Qm, W1p, b1, W2p, b2, W3, b3, dT, cT);

    // 3) softmax over n + score
    softmax_score_kernel<<<MM, 256, 0, stream>>>(dT, cT, out);
}
